// TokenBottleneck_7541962572349
// MI455X (gfx1250) — matrix operand order repaired and checked by reading, not compiled, not run on hardware
//
#include <hip/hip_runtime.h>
#include <hip/hip_bf16.h>
#include <math.h>

// ---------------------------------------------------------------------------
// TokenBottleneck for MI455X (gfx1250, wave32, WMMA)
// tokens[B,N,H] -> LayerNorm -> relu(x@w1+b1)@w2+b2 -> masked scores ->
// top-k(1024) -> gather.
// B=8, N=8192, H=1024, S=256, k=1024.
// ---------------------------------------------------------------------------

typedef __attribute__((ext_vector_type(16))) __bf16 v16bf;
typedef __attribute__((ext_vector_type(8)))  float  v8f;

static constexpr int BATCH  = 8;
static constexpr int NTOK   = 8192;
static constexpr int HDIM   = 1024;   // K of GEMM-1
static constexpr int SDIM   = 256;    // N of GEMM-1
static constexpr int TOPK   = 1024;
static constexpr int KTILES = HDIM / 32;   // 32 K-tiles of 32
static constexpr int NTILES = SDIM / 16;   // 16 N-tiles of 16
static constexpr int FRAG_DW = NTILES * 32 * 8;  // dwords of B-frags per K-tile (4096)
static constexpr float LN_EPS = 1e-5f;

__device__ inline unsigned bfpack2(float lo, float hi) {
  // round-to-nearest-even f32 -> bf16, pack two into a dword
  unsigned ul = __float_as_uint(lo);
  unsigned uh = __float_as_uint(hi);
  ul = (ul + 0x7FFFu + ((ul >> 16) & 1u)) >> 16;
  uh = (uh + 0x7FFFu + ((uh >> 16) & 1u)) >> 16;
  return (ul & 0xFFFFu) | (uh << 16);
}

union FragBF16 { unsigned u[8]; uint4 q[2]; v16bf v; };

// Async global->LDS copy of one 64-byte slice (4 x b128), tracked by ASYNCcnt.
// Per ISA 10.7/15.18: INST_OFFSET is added to BOTH the LDS and global address.
__device__ inline void async_copy64(unsigned lds_addr, const unsigned* gaddr) {
  asm volatile(
      "global_load_async_to_lds_b128 %0, %1, off\n\t"
      "global_load_async_to_lds_b128 %0, %1, off offset:16\n\t"
      "global_load_async_to_lds_b128 %0, %1, off offset:32\n\t"
      "global_load_async_to_lds_b128 %0, %1, off offset:48"
      :: "v"(lds_addr), "v"(gaddr) : "memory");
}
__device__ inline void wait_async0() {
  asm volatile("s_wait_asynccnt 0" ::: "memory");
}

// ---------------------------------------------------------------------------
// Kernel 1: repack w1 [H=1024][S=256] f32 -> bf16 WMMA B-fragments.
// Layout per (kt, nt) fragment (32x16, per ISA 16-bit B layout):
//   lane 0..15  : col n = lane,     K = kt*32 + {0..7, 16..23}  (2 per VGPR)
//   lane 16..31 : col n = lane-16,  K = kt*32 + {8..15, 24..31}
// Stored so a lane loads its 8 B-regs as 2 contiguous uint4.
// ---------------------------------------------------------------------------
__global__ void tb_prep_w1(const float* __restrict__ w1, unsigned* __restrict__ frag) {
  const int fid  = blockIdx.x;            // kt*NTILES + nt
  const int kt   = fid >> 4;
  const int nt   = fid & 15;
  const int lane = threadIdx.x & 31;
  const int n    = nt * 16 + (lane & 15);
  const int kb   = kt * 32 + ((lane >> 4) << 3);   // +8 for upper lane half
  unsigned* out = frag + ((size_t)fid * 32 + lane) * 8;
#pragma unroll
  for (int v = 0; v < 8; ++v) {
    const int k0 = kb + 2 * v + (v >= 4 ? 8 : 0);
    out[v] = bfpack2(w1[(size_t)k0 * SDIM + n], w1[(size_t)(k0 + 1) * SDIM + n]);
  }
}

// ---------------------------------------------------------------------------
// Kernel 2: LayerNorm + scorer MLP via v_wmma_f32_16x16x32_bf16.
// One wave (32 lanes) = one 16-token tile; 8 waves/block share B-fragments
// staged per K-tile into double-buffered LDS via async global->LDS copies.
// ---------------------------------------------------------------------------
__global__ void __launch_bounds__(256)
tb_score(const float* __restrict__ tokens,
         const float* __restrict__ gamma, const float* __restrict__ beta,
         const unsigned* __restrict__ w1frag,
         const float* __restrict__ b1, const float* __restrict__ w2,
         const float* __restrict__ b2, const int* __restrict__ mask,
         float* __restrict__ scores_out) {
  __shared__ unsigned bsm[2][FRAG_DW];   // 2 x 16 KB B-fragment buffers

  const int t    = threadIdx.x;
  const int lane = t & 31;
  const int tile = blockIdx.x * 8 + (t >> 5);  // 16-token tile id
  const int m    = lane & 15;        // row within tile owned by this lane
  const int hi   = lane >> 4;        // lane half
  const float* __restrict__ row = tokens + ((size_t)tile * 16 + m) * HDIM;

  // ---- Pass 1: LayerNorm statistics (each lane: 512 cols of its row) ----
  // Kick off the K-tile-0 fragment copy first so it overlaps the stats pass.
  {
    const unsigned lds0 = (unsigned)(uintptr_t)&bsm[0][t * 16];
    async_copy64(lds0, w1frag + (size_t)t * 16);
  }
  float s = 0.f, s2 = 0.f;
  {
    const float4* p = (const float4*)(row + hi * 512);
#pragma unroll 8
    for (int i = 0; i < 128; ++i) {
      float4 v = p[i];
      s  += v.x + v.y + v.z + v.w;
      s2 += v.x * v.x + v.y * v.y + v.z * v.z + v.w * v.w;
    }
  }
  s  += __shfl_xor(s, 16, 32);
  s2 += __shfl_xor(s2, 16, 32);
  const float mu  = s * (1.f / HDIM);
  const float var = s2 * (1.f / HDIM) - mu * mu;
  const float rs  = rsqrtf(var + LN_EPS);

  wait_async0();
  __syncthreads();

  // ---- Pass 2: GEMM-1, D[16x16] per N-tile, accumulate over 32 K-tiles ----
  v8f acc[NTILES];
#pragma unroll
  for (int nt = 0; nt < NTILES; ++nt)
    acc[nt] = (v8f){0.f, 0.f, 0.f, 0.f, 0.f, 0.f, 0.f, 0.f};

  const int ko = hi << 3;  // ISA 16-bit A layout: lane half offsets K by 8
  for (int kt = 0; kt < KTILES; ++kt) {
    const int cur = kt & 1;
    // Stage next K-tile's B fragments (16 KB/block) while we compute.
    // Safe: the barrier ending iteration kt-1 released all readers of buf 1-cur.
    if (kt + 1 < KTILES) {
      const unsigned ldsn = (unsigned)(uintptr_t)&bsm[cur ^ 1][t * 16];
      async_copy64(ldsn, w1frag + (size_t)(kt + 1) * FRAG_DW + t * 16);
    }

    const int c0 = kt * 32 + ko;        // first 8-col run
    const int c1 = c0 + 16;             // second 8-col run
    if (kt + 1 < KTILES)                // global_prefetch next token K-tile
      __builtin_prefetch(row + c0 + 32, 0, 1);

    float4 x0 = *(const float4*)(row + c0);
    float4 x1 = *(const float4*)(row + c0 + 4);
    float4 x2 = *(const float4*)(row + c1);
    float4 x3 = *(const float4*)(row + c1 + 4);
    float4 g0 = *(const float4*)(gamma + c0);
    float4 g1 = *(const float4*)(gamma + c0 + 4);
    float4 g2 = *(const float4*)(gamma + c1);
    float4 g3 = *(const float4*)(gamma + c1 + 4);
    float4 e0 = *(const float4*)(beta + c0);
    float4 e1 = *(const float4*)(beta + c0 + 4);
    float4 e2 = *(const float4*)(beta + c1);
    float4 e3 = *(const float4*)(beta + c1 + 4);

    float y[16];
    y[0]=(x0.x-mu)*rs*g0.x+e0.x; y[1]=(x0.y-mu)*rs*g0.y+e0.y;
    y[2]=(x0.z-mu)*rs*g0.z+e0.z; y[3]=(x0.w-mu)*rs*g0.w+e0.w;
    y[4]=(x1.x-mu)*rs*g1.x+e1.x; y[5]=(x1.y-mu)*rs*g1.y+e1.y;
    y[6]=(x1.z-mu)*rs*g1.z+e1.z; y[7]=(x1.w-mu)*rs*g1.w+e1.w;
    y[8]=(x2.x-mu)*rs*g2.x+e2.x; y[9]=(x2.y-mu)*rs*g2.y+e2.y;
    y[10]=(x2.z-mu)*rs*g2.z+e2.z; y[11]=(x2.w-mu)*rs*g2.w+e2.w;
    y[12]=(x3.x-mu)*rs*g3.x+e3.x; y[13]=(x3.y-mu)*rs*g3.y+e3.y;
    y[14]=(x3.z-mu)*rs*g3.z+e3.z; y[15]=(x3.w-mu)*rs*g3.w+e3.w;

    FragBF16 A;
#pragma unroll
    for (int v = 0; v < 4; ++v) {
      A.u[v]     = bfpack2(y[2 * v],     y[2 * v + 1]);      // K = ko+2v
      A.u[4 + v] = bfpack2(y[8 + 2 * v], y[8 + 2 * v + 1]);  // K = 16+ko+2v
    }

    // B from LDS: frag nt = uint4 [nt*64, nt*64+64), lane slice = 2 uint4.
    const uint4* bf = (const uint4*)&bsm[cur][0];
#pragma unroll
    for (int nt = 0; nt < NTILES; ++nt) {
      FragBF16 Bm;
      Bm.q[0] = bf[nt * 64 + lane * 2 + 0];
      Bm.q[1] = bf[nt * 64 + lane * 2 + 1];
      acc[nt] = __builtin_amdgcn_wmma_f32_16x16x32_bf16(
          /*neg_a=*/false, A.v, /*neg_b=*/false, Bm.v,
          /*c_mod=*/(short)0, acc[nt], /*reuse_a=*/false, /*reuse_b=*/false);
    }

    wait_async0();      // next buffer landed
    __syncthreads();    // all waves done reading cur, next buffer visible
  }

  // ---- bias + ReLU + dot(w2): D-layout lane holds (M = r + 8*hi, n = lane&15)
  const int n0 = lane & 15;
  float p[8] = {0.f, 0.f, 0.f, 0.f, 0.f, 0.f, 0.f, 0.f};
#pragma unroll
  for (int nt = 0; nt < NTILES; ++nt) {
    const float b1n = b1[nt * 16 + n0];
    const float w2n = w2[nt * 16 + n0];
#pragma unroll
    for (int r = 0; r < 8; ++r) {
      float h = acc[nt][r] + b1n;
      h = h > 0.f ? h : 0.f;
      p[r] += h * w2n;
    }
  }
#pragma unroll
  for (int off = 1; off < 16; off <<= 1) {
#pragma unroll
    for (int r = 0; r < 8; ++r) p[r] += __shfl_xor(p[r], off, 32);
  }
  if (n0 == 0) {  // lanes 0 and 16 own rows 0..7 / 8..15
    const float bias2 = b2[0];
    const int mbase = tile * 16 + hi * 8;
#pragma unroll
    for (int r = 0; r < 8; ++r) {
      const int tok = mbase + r;                 // == b*NTOK + n
      const bool mk = mask[tok] != 0;
      scores_out[tok] = mk ? (p[r] + bias2) : -INFINITY;
    }
  }
}

// ---------------------------------------------------------------------------
// Kernel 3: per-batch top-1024 by full bitonic sort of 8192 64-bit keys in LDS.
// key = (order-preserving score bits << 32) | (8191 - index)
//   -> descending sort: higher score first, lower index first on ties,
//      matching jax.lax.top_k.
// ---------------------------------------------------------------------------
__global__ void __launch_bounds__(1024)
tb_topk(const float* __restrict__ scores, const int* __restrict__ mask,
        int* __restrict__ indices_out, float* __restrict__ selmask_out) {
  __shared__ unsigned long long keys[NTOK];   // 64 KB of 320 KB WGP LDS
  const int b = blockIdx.x;
  const int tid = threadIdx.x;

  for (int i = tid; i < NTOK; i += 1024) {
    unsigned u = __float_as_uint(scores[(size_t)b * NTOK + i]);
    u = (u >> 31) ? ~u : (u | 0x80000000u);   // monotone float -> uint map
    keys[i] = ((unsigned long long)u << 32) | (unsigned)(NTOK - 1 - i);
  }
  __syncthreads();

  for (int k = 2; k <= NTOK; k <<= 1) {
    for (int j = k >> 1; j > 0; j >>= 1) {
      for (int i = tid; i < NTOK; i += 1024) {
        const int ixj = i ^ j;
        if (ixj > i) {
          const bool up = (i & k) == 0;
          unsigned long long a = keys[i], c = keys[ixj];
          const bool sw = up ? (a < c) : (a > c);   // descending overall
          if (sw) { keys[i] = c; keys[ixj] = a; }
        }
      }
      __syncthreads();
    }
  }

  if (tid < TOPK) {
    const unsigned long long key = keys[tid];
    const int idx = NTOK - 1 - (int)(key & 0xFFFFFFFFull);
    indices_out[b * TOPK + tid] = idx;
    const bool mk = mask[(size_t)b * NTOK + idx] != 0;
    selmask_out[b * TOPK + tid] = mk ? 1.f : 0.f;
  }
}

// ---------------------------------------------------------------------------
// Kernel 4: gather selected tokens (zero rows whose mask is false).
// One block per output row; float4 copies (4 per thread -> one b128 each).
// ---------------------------------------------------------------------------
__global__ void __launch_bounds__(256)
tb_gather(const float* __restrict__ tokens, const int* __restrict__ indices,
          const int* __restrict__ mask, float* __restrict__ gathered) {
  const int bi = blockIdx.x;          // b*TOPK + i
  const int b  = bi >> 10;
  int idx = indices[bi] & (NTOK - 1);
  const bool mk = mask[(size_t)b * NTOK + idx] != 0;
  const float4* src = (const float4*)(tokens + ((size_t)b * NTOK + idx) * HDIM);
  float4* dst = (float4*)(gathered + (size_t)bi * HDIM);
  const float4 z = {0.f, 0.f, 0.f, 0.f};
  dst[threadIdx.x] = mk ? src[threadIdx.x] : z;
}

// ---------------------------------------------------------------------------
extern "C" void kernel_launch(void* const* d_in, const int* in_sizes, int n_in,
                              void* d_out, int out_size, void* d_ws, size_t ws_size,
                              hipStream_t stream) {
  (void)in_sizes; (void)n_in; (void)out_size; (void)ws_size;
  const float* tokens = (const float*)d_in[0];
  const float* gamma  = (const float*)d_in[1];
  const float* beta   = (const float*)d_in[2];
  const float* w1     = (const float*)d_in[3];
  const float* b1     = (const float*)d_in[4];
  const float* w2     = (const float*)d_in[5];
  const float* b2     = (const float*)d_in[6];
  const int*   mask   = (const int*)d_in[7];
  // d_in[8] = budget (1024; grid shapes below assume it)

  float* out      = (float*)d_out;
  float* gathered = out;                                        // [8,1024,1024]
  int*   indices  = (int*)(out + (size_t)BATCH * TOPK * HDIM);  // [8,1024]
  float* scores   = out + (size_t)BATCH * TOPK * HDIM + BATCH * TOPK; // [8,8192]
  float* selmask  = scores + (size_t)BATCH * NTOK;              // [8,1024]
  unsigned* w1frag = (unsigned*)d_ws;                           // 512 KB

  tb_prep_w1<<<KTILES * NTILES, 32, 0, stream>>>(w1, w1frag);
  tb_score<<<(BATCH * NTOK) / (16 * 8), 256, 0, stream>>>(
      tokens, gamma, beta, w1frag, b1, w2, b2, mask, scores);
  tb_topk<<<BATCH, 1024, 0, stream>>>(scores, mask, indices, selmask);
  tb_gather<<<BATCH * TOPK, 256, 0, stream>>>(tokens, indices, mask, gathered);
}
